// GETLayer_37331855737112
// MI455X (gfx1250) — matrix operand order repaired and checked by reading, not compiled, not run on hardware
//
#include <hip/hip_runtime.h>
#include <math.h>

typedef _Float16 f16;
typedef __attribute__((ext_vector_type(16))) _Float16 v16h;
typedef __attribute__((ext_vector_type(8)))  _Float16 v8h;
typedef __attribute__((ext_vector_type(4)))  _Float16 v4h;
typedef __attribute__((ext_vector_type(8)))  float    v8f;

#define NN   32768
#define DD   128
#define HH   2
#define HDV  64
#define RR   2
#define KSL  32
#define EE   262144
#define MMOT 131072
#define NGG  32
#define CP   896          // concat cols: Q2[0:128) K2[128:256) Q3[256:512) K3[512:768) Qm[768:896)
#define C_Q2 0
#define C_K2 128
#define C_Q3 256
#define C_K3 512
#define C_QM 768
#define EPSV 1e-5f
// softplus(inv_softplus(x)) == x exactly:
#define LAM2 1.0f
#define LAM3 0.5f
#define LAMM 1.0f
// b2=b3=bm = min(softplus(isp(1)),5) = 1

// ---- order-preserving float <-> u32 map for atomicMax-based segment max ----
__device__ __forceinline__ unsigned fenc(float f) {
    unsigned u = __float_as_uint(f);
    return (u & 0x80000000u) ? ~u : (u | 0x80000000u);
}
__device__ __forceinline__ float fdec(unsigned u) {
    return (u & 0x80000000u) ? __uint_as_float(u & 0x7fffffffu) : __uint_as_float(~u);
}

__device__ __forceinline__ float wsum(float v) {
    #pragma unroll
    for (int o = 16; o > 0; o >>= 1) v += __shfl_xor(v, o, 32);
    return v;
}
__device__ __forceinline__ float wmax(float v) {
    #pragma unroll
    for (int o = 16; o > 0; o >>= 1) v = fmaxf(v, __shfl_xor(v, o, 32));
    return v;
}

// ---------------- weight prep: Wcat f16 [896,128] and WcatT f16 [128,896] ----------------
__global__ void k_prep_weights(const float* __restrict__ WQ2, const float* __restrict__ WK2,
                               const float* __restrict__ WQ3, const float* __restrict__ WK3,
                               const float* __restrict__ WQm,
                               f16* __restrict__ Wcat, f16* __restrict__ WcatT) {
    int idx = blockIdx.x * blockDim.x + threadIdx.x;
    if (idx >= CP * DD) return;
    int c = idx / DD, d = idx % DD;
    float w;
    if      (c < 256) w = (c < 128) ? WQ2[c * DD + d] : WK2[(c - 128) * DD + d];
    else if (c < 768) w = (c < 512) ? WQ3[(c - 256) * DD + d] : WK3[(c - 512) * DD + d];
    else              w = WQm[(c - 768) * DD + d];
    f16 h = (f16)w;
    Wcat[c * DD + d]  = h;
    WcatT[d * CP + c] = h;
}

// ---------------- Km[h,k,z] = sum_d B_mem[k,d] * W_Km[h,z,d] ----------------
__global__ void k_km(const float* __restrict__ Bm, const float* __restrict__ WKm,
                     float* __restrict__ Km) {
    int idx = blockIdx.x * blockDim.x + threadIdx.x;   // H*K*HD = 4096
    if (idx >= HH * KSL * HDV) return;
    int z = idx % HDV, k = (idx / HDV) % KSL, h = idx / (KSL * HDV);
    const float4* b = (const float4*)(Bm + (size_t)k * DD);
    const float4* w = (const float4*)(WKm + (size_t)(h * HDV + z) * DD);
    float s = 0.f;
    #pragma unroll
    for (int i = 0; i < DD / 4; ++i) {
        float4 a = b[i], c = w[i];
        s += a.x * c.x + a.y * c.y + a.z * c.z + a.w * c.w;
    }
    Km[idx] = s;
}

// ---------------- layernorm fwd: Gh (f16), mu, rstd ----------------
__global__ void k_layernorm(const float* __restrict__ X, const float* __restrict__ gam,
                            const float* __restrict__ bet, f16* __restrict__ Gh,
                            float* __restrict__ mu_o, float* __restrict__ rstd_o) {
    int lane = threadIdx.x & 31;
    int row  = blockIdx.x * (blockDim.x >> 5) + (threadIdx.x >> 5);
    const float4 x4 = ((const float4*)(X + (size_t)row * DD))[lane];
    float mu = wsum(x4.x + x4.y + x4.z + x4.w) * (1.0f / DD);
    float d0 = x4.x - mu, d1 = x4.y - mu, d2 = x4.z - mu, d3 = x4.w - mu;
    float var = wsum(d0 * d0 + d1 * d1 + d2 * d2 + d3 * d3) * (1.0f / DD);
    float rstd = rsqrtf(var + EPSV);
    const float4 g4 = ((const float4*)gam)[lane];
    const float4 b4 = ((const float4*)bet)[lane];
    v4h o;
    o[0] = (f16)(g4.x * d0 * rstd + b4.x);
    o[1] = (f16)(g4.y * d1 * rstd + b4.y);
    o[2] = (f16)(g4.z * d2 * rstd + b4.z);
    o[3] = (f16)(g4.w * d3 * rstd + b4.w);
    ((v4h*)(Gh + (size_t)row * DD))[lane] = o;
    if (lane == 0) { mu_o[row] = mu; rstd_o[row] = rstd; }
}

// ---- forward projection: P[N,896] = Gh[N,128] x Wcat^T via WMMA, 16x64 per wave ----
__global__ void k_proj_fwd(const f16* __restrict__ Gh, const f16* __restrict__ Wcat,
                           float* __restrict__ P) {
    int lane = threadIdx.x & 31;
    int wv   = blockIdx.x * (blockDim.x >> 5) + (threadIdx.x >> 5);
    const int CG = CP / 64;                     // 14 groups of 4 col tiles
    int tm = wv / CG, tg = wv % CG;
    int cb = tg * 64;
    int l = lane & 15, hi = lane >> 4;
    const f16* arow = Gh + (size_t)(tm * 16 + l) * DD;   // A: row M=l, 16x32 f16 layout
    v8f zero = {0.f, 0.f, 0.f, 0.f, 0.f, 0.f, 0.f, 0.f};
    v8f acc[4] = {zero, zero, zero, zero};
    #pragma unroll
    for (int kk = 0; kk < DD / 32; ++kk) {
        int ka = kk * 32 + hi * 8;              // lanes0-15: K0-7,K16-23 ; lanes16-31: K8-15,K24-31
        v8h a0 = *(const v8h*)(arow + ka);
        v8h a1 = *(const v8h*)(arow + ka + 16);
        v16h a;
        #pragma unroll
        for (int i = 0; i < 8; ++i) { a[i] = a0[i]; a[i + 8] = a1[i]; }
        int kb = kk * 32 + hi * 8;              // lanes0-15: K0-7,K16-23 ; lanes16-31: K8-15,K24-31
        #pragma unroll
        for (int j = 0; j < 4; ++j) {           // reuse A across 4 col tiles (independent accs)
            const f16* brow = Wcat + (size_t)(cb + j * 16 + l) * DD;
            v8h b0 = *(const v8h*)(brow + kb);
            v8h b1 = *(const v8h*)(brow + kb + 16);
            v16h b;
            #pragma unroll
            for (int i = 0; i < 8; ++i) { b[i] = b0[i]; b[i + 8] = b1[i]; }
            acc[j] = __builtin_amdgcn_wmma_f32_16x16x32_f16(false, a, false, b, (short)0,
                                                            acc[j], false, false);
        }
    }
    int r0 = tm * 16 + hi * 8;
    #pragma unroll
    for (int j = 0; j < 4; ++j) {
        int c = cb + j * 16 + l;
        #pragma unroll
        for (int jj = 0; jj < 8; ++jj) P[(size_t)(r0 + jj) * CP + c] = acc[j][jj];
    }
}

// ---- backward projection: dG[N,128] = dP[N,896] x Wcat via WMMA, 16x64 per wave ----
__global__ void k_proj_bwd(const float* __restrict__ dP, const f16* __restrict__ WcatT,
                           float* __restrict__ dG) {
    int lane = threadIdx.x & 31;
    int wv   = blockIdx.x * (blockDim.x >> 5) + (threadIdx.x >> 5);
    const int CG = DD / 64;                     // 2 groups of 4 col tiles
    int tm = wv / CG, tg = wv % CG;
    int cb = tg * 64;
    int l = lane & 15, hi = lane >> 4;
    const float* arow = dP + (size_t)(tm * 16 + l) * CP;
    v8f zero = {0.f, 0.f, 0.f, 0.f, 0.f, 0.f, 0.f, 0.f};
    v8f acc[4] = {zero, zero, zero, zero};
    for (int kk = 0; kk < CP / 32; ++kk) {      // 28 K-steps
        int ka = kk * 32 + hi * 8;
        float4 f0 = *(const float4*)(arow + ka);
        float4 f1 = *(const float4*)(arow + ka + 4);
        float4 f2 = *(const float4*)(arow + ka + 16);
        float4 f3 = *(const float4*)(arow + ka + 20);
        v16h a;
        a[0]  = (f16)f0.x; a[1]  = (f16)f0.y; a[2]  = (f16)f0.z; a[3]  = (f16)f0.w;
        a[4]  = (f16)f1.x; a[5]  = (f16)f1.y; a[6]  = (f16)f1.z; a[7]  = (f16)f1.w;
        a[8]  = (f16)f2.x; a[9]  = (f16)f2.y; a[10] = (f16)f2.z; a[11] = (f16)f2.w;
        a[12] = (f16)f3.x; a[13] = (f16)f3.y; a[14] = (f16)f3.z; a[15] = (f16)f3.w;
        int kb = kk * 32 + hi * 8;
        #pragma unroll
        for (int j = 0; j < 4; ++j) {
            const f16* brow = WcatT + (size_t)(cb + j * 16 + l) * CP;
            v8h b0 = *(const v8h*)(brow + kb);
            v8h b1 = *(const v8h*)(brow + kb + 16);
            v16h b;
            #pragma unroll
            for (int i = 0; i < 8; ++i) { b[i] = b0[i]; b[i + 8] = b1[i]; }
            acc[j] = __builtin_amdgcn_wmma_f32_16x16x32_f16(false, a, false, b, (short)0,
                                                            acc[j], false, false);
        }
    }
    int r0 = tm * 16 + hi * 8;
    #pragma unroll
    for (int j = 0; j < 4; ++j) {
        int c = cb + j * 16 + l;
        #pragma unroll
        for (int jj = 0; jj < 8; ++jj) dG[(size_t)(r0 + jj) * DD + c] = acc[j][jj];
    }
}

// ---------------- edge scores + segment max ----------------
__global__ void k_edge_score(const int* __restrict__ c2, const int* __restrict__ u2,
                             const float* __restrict__ a2, const float* __restrict__ P,
                             float* __restrict__ s2, unsigned* __restrict__ m2e) {
    int idx = blockIdx.x * blockDim.x + threadIdx.x;   // E*H
    if (idx >= EE * HH) return;
    int e = idx >> 1, h = idx & 1;
    int c = c2[e], u = u2[e];
    const float4* q = (const float4*)(P + (size_t)c * CP + C_Q2 + h * HDV);
    const float4* k = (const float4*)(P + (size_t)u * CP + C_K2 + h * HDV);
    float s = 0.f;
    #pragma unroll
    for (int i = 0; i < HDV / 4; ++i) {
        float4 a = q[i], b = k[i];
        s += a.x * b.x + a.y * b.y + a.z * b.z + a.w * b.w;
    }
    s = s * 0.125f + a2[e * HH + h];            // /sqrt(64)
    s2[idx] = s;
    atomicMax(m2e + c * HH + h, fenc(s));       // beta2 = 1
}

// ---------------- motif scores + segment max (stores qk, tv for bwd) ----------------
__global__ void k_motif_score(const int* __restrict__ c3, const int* __restrict__ u3,
                              const int* __restrict__ v3, const int* __restrict__ tt,
                              const float* __restrict__ T, const float* __restrict__ P,
                              float* __restrict__ s3, float* __restrict__ qk3,
                              float* __restrict__ tv3, unsigned* __restrict__ m3e) {
    int idx = blockIdx.x * blockDim.x + threadIdx.x;   // M*H
    if (idx >= MMOT * HH) return;
    int m = idx >> 1, h = idx & 1;
    int c = c3[m], u = u3[m], v = v3[m], t = tt[m];
    float sv = 0.f;
    #pragma unroll
    for (int r = 0; r < RR; ++r) {
        const float4* q  = (const float4*)(P + (size_t)c * CP + C_Q3 + h * (RR * HDV) + r * HDV);
        const float4* ku = (const float4*)(P + (size_t)u * CP + C_K3 + h * (RR * HDV) + r * HDV);
        const float4* kv = (const float4*)(P + (size_t)v * CP + C_K3 + h * (RR * HDV) + r * HDV);
        const float4* tw = (const float4*)(T + (size_t)((t * HH + h) * RR + r) * HDV);
        float qk = 0.f, tv = 0.f;
        #pragma unroll
        for (int i = 0; i < HDV / 4; ++i) {
            float4 a = q[i], b = ku[i], x = tw[i], y = kv[i];
            qk += a.x * b.x + a.y * b.y + a.z * b.z + a.w * b.w;
            tv += x.x * y.x + x.y * y.y + x.z * y.z + x.w * y.w;
        }
        qk3[idx * RR + r] = qk;
        tv3[idx * RR + r] = tv;
        sv += qk * tv;
    }
    sv *= (1.0f / HDV);
    s3[idx] = sv;
    atomicMax(m3e + c * HH + h, fenc(sv));      // beta3 = 1
}

// ---------------- softmax denominators ----------------
__global__ void k_edge_z(const int* __restrict__ c2, const float* __restrict__ s2,
                         const unsigned* __restrict__ m2e, float* __restrict__ z2) {
    int idx = blockIdx.x * blockDim.x + threadIdx.x;
    if (idx >= EE * HH) return;
    int e = idx >> 1, h = idx & 1, c = c2[e];
    atomicAdd(z2 + c * HH + h, __expf(s2[idx] - fdec(m2e[c * HH + h])));
}
__global__ void k_motif_z(const int* __restrict__ c3, const float* __restrict__ s3,
                          const unsigned* __restrict__ m3e, float* __restrict__ z3) {
    int idx = blockIdx.x * blockDim.x + threadIdx.x;
    if (idx >= MMOT * HH) return;
    int m = idx >> 1, h = idx & 1, c = c3[m];
    atomicAdd(z3 + c * HH + h, __expf(s3[idx] - fdec(m3e[c * HH + h])));
}

// ---------------- per-node: memory term (energy + dQm) + node energy -> Eg ----------------
__global__ void k_node(const float* __restrict__ P, const float* __restrict__ Km,
                       const float* __restrict__ mu, const float* __restrict__ rstd,
                       const unsigned* __restrict__ m2e, const float* __restrict__ z2,
                       const unsigned* __restrict__ m3e, const float* __restrict__ z3,
                       const int* __restrict__ batch, float* __restrict__ dP,
                       float* __restrict__ Eg) {
    int lane = threadIdx.x & 31;                      // lane == memory slot k (K==32)
    int n    = blockIdx.x * (blockDim.x >> 5) + (threadIdx.x >> 5);
    float e_mem = 0.f;
    #pragma unroll
    for (int h = 0; h < HH; ++h) {
        const float4* qm = (const float4*)(P + (size_t)n * CP + C_QM + h * HDV);
        const float4* km = (const float4*)(Km + (size_t)(h * KSL + lane) * HDV);
        float s = 0.f;
        #pragma unroll
        for (int i = 0; i < HDV / 4; ++i) {
            float4 a = qm[i], b = km[i];
            s += a.x * b.x + a.y * b.y + a.z * b.z + a.w * b.w;
        }
        s *= 0.125f;                                  // sm[n,h,k]
        float mx = wmax(s);
        float ex = __expf(s - mx);
        float zz = wsum(ex);
        float p  = ex / zz;                           // softmax over k (bm=1)
        e_mem   -= (mx + __logf(zz));                 // -(lamm/bm)*lse
        float* dqm = dP + (size_t)n * CP + C_QM + h * HDV;
        #pragma unroll
        for (int i = 0; i < HDV / 4; ++i) {
            float4 kv = km[i];
            float cx = wsum(p * kv.x), cy = wsum(p * kv.y);
            float cz = wsum(p * kv.z), cw = wsum(p * kv.w);
            if (lane == 0) {
                float4 o; o.x = -0.125f * cx; o.y = -0.125f * cy;
                o.z = -0.125f * cz; o.w = -0.125f * cw;
                ((float4*)dqm)[i] = o;                // dQm = -lamm/sqrt(HD) * sum_k p_k Km
            }
        }
    }
    if (lane == 0) {
        float rs = rstd[n];
        float var = 1.0f / (rs * rs) - EPSV;
        float e = 0.5f * DD * (var + mu[n] * mu[n]);  // 0.5*||x||^2
        #pragma unroll
        for (int h = 0; h < HH; ++h) {
            float zp = z2[n * HH + h];
            if (zp > 0.f) e -= LAM2 * (fdec(m2e[n * HH + h]) + __logf(fmaxf(zp, 1e-30f)));
            float zm = z3[n * HH + h];
            if (zm > 0.f) e -= LAM3 * (fdec(m3e[n * HH + h]) + __logf(fmaxf(zm, 1e-30f)));
        }
        e += e_mem;
        atomicAdd(Eg + batch[n], e);
    }
}

// ---------------- edge backward: scatter into dQ2 / dK2 ----------------
__global__ void k_edge_bwd(const int* __restrict__ c2, const int* __restrict__ u2,
                           const float* __restrict__ s2, const unsigned* __restrict__ m2e,
                           const float* __restrict__ z2, const float* __restrict__ P,
                           float* __restrict__ dP) {
    int lane = threadIdx.x & 31;
    int e    = blockIdx.x * (blockDim.x >> 5) + (threadIdx.x >> 5);
    int c = c2[e], u = u2[e];
    #pragma unroll
    for (int h = 0; h < HH; ++h) {
        float zp = z2[c * HH + h];
        if (zp <= 0.f) continue;
        float p    = __expf(s2[e * HH + h] - fdec(m2e[c * HH + h])) / zp;
        float coef = -LAM2 * p * 0.125f;              // -lam2 * p / sqrt(HD)
        const float* q = P + (size_t)c * CP + C_Q2 + h * HDV;
        const float* k = P + (size_t)u * CP + C_K2 + h * HDV;
        float* dq = dP + (size_t)c * CP + C_Q2 + h * HDV;
        float* dk = dP + (size_t)u * CP + C_K2 + h * HDV;
        int z = lane * 2;
        atomicAdd(dq + z,     coef * k[z]);
        atomicAdd(dq + z + 1, coef * k[z + 1]);
        atomicAdd(dk + z,     coef * q[z]);
        atomicAdd(dk + z + 1, coef * q[z + 1]);
    }
}

// ---------------- motif backward: scatter into dQ3 / dK3 ----------------
__global__ void k_motif_bwd(const int* __restrict__ c3, const int* __restrict__ u3,
                            const int* __restrict__ v3, const int* __restrict__ tt,
                            const float* __restrict__ s3, const float* __restrict__ qk3,
                            const float* __restrict__ tv3, const unsigned* __restrict__ m3e,
                            const float* __restrict__ z3, const float* __restrict__ T,
                            const float* __restrict__ P, float* __restrict__ dP) {
    int lane = threadIdx.x & 31;
    int m    = blockIdx.x * (blockDim.x >> 5) + (threadIdx.x >> 5);
    int c = c3[m], u = u3[m], v = v3[m], t = tt[m];
    int hr = lane >> 3, h = hr >> 1, r = hr & 1, z0 = (lane & 7) * 8;
    float zp = z3[c * HH + h];
    float p = 0.f;
    if (zp > 0.f) p = __expf(s3[m * HH + h] - fdec(m3e[c * HH + h])) / zp;
    float g  = -LAM3 * p * (1.0f / HDV);              // d e_motif / d s3 scaled by 1/HD
    float cq = g * tv3[(m * HH + h) * RR + r];
    float cv = g * qk3[(m * HH + h) * RR + r];
    const float* q  = P + (size_t)c * CP + C_Q3 + h * (RR * HDV) + r * HDV;
    const float* ku = P + (size_t)u * CP + C_K3 + h * (RR * HDV) + r * HDV;
    const float* tw = T + (size_t)((t * HH + h) * RR + r) * HDV;
    float* dq  = dP + (size_t)c * CP + C_Q3 + h * (RR * HDV) + r * HDV;
    float* dku = dP + (size_t)u * CP + C_K3 + h * (RR * HDV) + r * HDV;
    float* dkv = dP + (size_t)v * CP + C_K3 + h * (RR * HDV) + r * HDV;
    #pragma unroll
    for (int j = 0; j < 8; ++j) {
        int z = z0 + j;
        atomicAdd(dq  + z, cq * ku[z]);
        atomicAdd(dku + z, cq * q[z]);
        atomicAdd(dkv + z, cv * tw[z]);
    }
}

// ---------------- final: LN backward + node term + clipped update ----------------
__global__ void k_final(const float* __restrict__ X, const float* __restrict__ dG,
                        const float* __restrict__ gam, const float* __restrict__ mu,
                        const float* __restrict__ rstd, const float* __restrict__ step,
                        float* __restrict__ Xo) {
    int lane = threadIdx.x & 31;
    int n    = blockIdx.x * (blockDim.x >> 5) + (threadIdx.x >> 5);
    const float4 x4 = ((const float4*)(X  + (size_t)n * DD))[lane];
    const float4 g4 = ((const float4*)(dG + (size_t)n * DD))[lane];
    const float4 a4 = ((const float4*)gam)[lane];
    float m = mu[n], rs = rstd[n];
    float4 xh; xh.x = (x4.x - m) * rs; xh.y = (x4.y - m) * rs;
    xh.z = (x4.z - m) * rs; xh.w = (x4.w - m) * rs;
    float4 dh; dh.x = g4.x * a4.x; dh.y = g4.y * a4.y;
    dh.z = g4.z * a4.z; dh.w = g4.w * a4.w;
    float s1 = wsum(dh.x + dh.y + dh.z + dh.w) * (1.0f / DD);
    float s2 = wsum(dh.x * xh.x + dh.y * xh.y + dh.z * xh.z + dh.w * xh.w) * (1.0f / DD);
    float4 gx;
    gx.x = x4.x + rs * (dh.x - s1 - xh.x * s2);
    gx.y = x4.y + rs * (dh.y - s1 - xh.y * s2);
    gx.z = x4.z + rs * (dh.z - s1 - xh.z * s2);
    gx.w = x4.w + rs * (dh.w - s1 - xh.w * s2);
    float gn = sqrtf(wsum(gx.x * gx.x + gx.y * gx.y + gx.z * gx.z + gx.w * gx.w));
    float sc = 1.0f / fmaxf(gn, 1.0f);                // GRAD_CLIP=1
    float st = step[0] * sc;                          // DAMPING=1
    float4 xn; xn.x = x4.x - st * gx.x; xn.y = x4.y - st * gx.y;
    xn.z = x4.z - st * gx.z; xn.w = x4.w - st * gx.w;
    float sn = sqrtf(wsum(xn.x * xn.x + xn.y * xn.y + xn.z * xn.z + xn.w * xn.w));
    float sc2 = 10.0f / fmaxf(sn, 10.0f);             // STATE_CLIP=10
    float4 o; o.x = xn.x * sc2; o.y = xn.y * sc2; o.z = xn.z * sc2; o.w = xn.w * sc2;
    ((float4*)(Xo + (size_t)n * DD))[lane] = o;
}

extern "C" void kernel_launch(void* const* d_in, const int* in_sizes, int n_in,
                              void* d_out, int out_size, void* d_ws, size_t ws_size,
                              hipStream_t stream) {
    const float* X    = (const float*)d_in[0];
    const int*   c2   = (const int*)d_in[1];
    const int*   u2   = (const int*)d_in[2];
    const int*   c3   = (const int*)d_in[3];
    const int*   u3   = (const int*)d_in[4];
    const int*   v3   = (const int*)d_in[5];
    const int*   tt   = (const int*)d_in[6];
    const int*   bat  = (const int*)d_in[7];
    const float* a2   = (const float*)d_in[8];
    const float* step = (const float*)d_in[9];
    const float* gam  = (const float*)d_in[10];
    const float* bet  = (const float*)d_in[11];
    const float* WQ2  = (const float*)d_in[12];
    const float* WK2  = (const float*)d_in[13];
    const float* WQ3  = (const float*)d_in[14];
    const float* WK3  = (const float*)d_in[15];
    const float* Ttau = (const float*)d_in[16];
    const float* WQm  = (const float*)d_in[17];
    const float* WKm  = (const float*)d_in[18];
    const float* Bm   = (const float*)d_in[19];

    float* out    = (float*)d_out;
    float* X_next = out;
    float* Eg     = out + (size_t)NN * DD;

    char*  ws  = (char*)d_ws;
    size_t off = 0;
    auto alloc = [&](size_t bytes) -> void* {
        size_t a = (off + 255) & ~(size_t)255;
        void* p  = ws + a;
        off = a + bytes;
        return p;
    };
    f16*      Gh    = (f16*)alloc((size_t)NN * DD * 2);
    f16*      Wcat  = (f16*)alloc((size_t)CP * DD * 2);
    f16*      WcatT = (f16*)alloc((size_t)DD * CP * 2);
    float*    muv   = (float*)alloc((size_t)NN * 4);
    float*    rstd  = (float*)alloc((size_t)NN * 4);
    float*    P     = (float*)alloc((size_t)NN * CP * 4);
    float*    dP    = (float*)alloc((size_t)NN * CP * 4);
    float*    Km    = (float*)alloc((size_t)HH * KSL * HDV * 4);
    float*    s2    = (float*)alloc((size_t)EE * HH * 4);
    float*    s3    = (float*)alloc((size_t)MMOT * HH * 4);
    float*    qk3   = (float*)alloc((size_t)MMOT * HH * RR * 4);
    float*    tv3   = (float*)alloc((size_t)MMOT * HH * RR * 4);
    unsigned* m2e   = (unsigned*)alloc((size_t)NN * HH * 4);
    float*    z2    = (float*)alloc((size_t)NN * HH * 4);
    unsigned* m3e   = (unsigned*)alloc((size_t)NN * HH * 4);
    float*    z3    = (float*)alloc((size_t)NN * HH * 4);
    float*    dG    = (float*)alloc((size_t)NN * DD * 4);
    (void)ws_size; (void)n_in; (void)in_sizes; (void)out_size;

    hipMemsetAsync(dP,  0, (size_t)NN * CP * 4, stream);
    hipMemsetAsync(m2e, 0, (size_t)NN * HH * 4, stream);
    hipMemsetAsync(z2,  0, (size_t)NN * HH * 4, stream);
    hipMemsetAsync(m3e, 0, (size_t)NN * HH * 4, stream);
    hipMemsetAsync(z3,  0, (size_t)NN * HH * 4, stream);
    hipMemsetAsync(Eg,  0, (size_t)NGG * 4, stream);

    k_prep_weights<<<(CP * DD + 255) / 256, 256, 0, stream>>>(WQ2, WK2, WQ3, WK3, WQm,
                                                              Wcat, WcatT);
    k_km<<<(HH * KSL * HDV + 255) / 256, 256, 0, stream>>>(Bm, WKm, Km);
    k_layernorm<<<NN / 8, 256, 0, stream>>>(X, gam, bet, Gh, muv, rstd);
    k_proj_fwd<<<(NN / 16) * (CP / 64) / 8, 256, 0, stream>>>(Gh, Wcat, P);
    k_edge_score<<<EE * HH / 256, 256, 0, stream>>>(c2, u2, a2, P, s2, m2e);
    k_motif_score<<<MMOT * HH / 256, 256, 0, stream>>>(c3, u3, v3, tt, Ttau, P, s3, qk3,
                                                       tv3, m3e);
    k_edge_z<<<EE * HH / 256, 256, 0, stream>>>(c2, s2, m2e, z2);
    k_motif_z<<<MMOT * HH / 256, 256, 0, stream>>>(c3, s3, m3e, z3);
    k_node<<<NN / 8, 256, 0, stream>>>(P, Km, muv, rstd, m2e, z2, m3e, z3, bat, dP, Eg);
    k_edge_bwd<<<EE / 8, 256, 0, stream>>>(c2, u2, s2, m2e, z2, P, dP);
    k_motif_bwd<<<MMOT / 8, 256, 0, stream>>>(c3, u3, v3, tt, s3, qk3, tv3, m3e, z3, Ttau,
                                              P, dP);
    k_proj_bwd<<<(NN / 16) * (DD / 64) / 8, 256, 0, stream>>>(dP, WcatT, dG);
    k_final<<<NN / 8, 256, 0, stream>>>(X, dG, gam, muv, rstd, step, X_next);
}
